// Head_29643864277606
// MI455X (gfx1250) — matrix operand order repaired and checked by reading, not compiled, not run on hardware
//
#include <hip/hip_runtime.h>
#include <hip/hip_bf16.h>

typedef _Float16 v16h __attribute__((ext_vector_type(16)));
typedef _Float16 v8h  __attribute__((ext_vector_type(8)));
typedef _Float16 v4h  __attribute__((ext_vector_type(4)));
typedef float    v8f  __attribute__((ext_vector_type(8)));

#define B_BATCH 4096
#define T_SEQ   124
#define C_EMB   144
#define H_DIM   36
#define TP      128   // padded T
#define CP      160   // padded C (5 K-chunks of 32)
#define WROWS   112   // padded 3*H columns of fused projection

// shared-memory layout, units of _Float16
#define XS_OFF  0        // x staging      128*160 = 20480 halfs
#define WT_OFF  20480    // W^T (f16)      112*160 = 17920
#define KB_OFF  38400    // K  [128][64]   = 8192
#define QB_OFF  46592    // Q  [128][64]   = 8192
#define VT_OFF  54784    // V^T[64][128]   = 8192
#define PB_OFF  0        // P  [128][128]  = 16384 (overlays XS, dead by then)
#define SH_HALFS 62976   // 125952 bytes total

__device__ __forceinline__ v16h ld_afrag(const _Float16* p) {
  // A 16x32 f16 fragment: halfs j=0..7 at k=8g.., j=8..15 at k=16+8g..
  union { v16h v; v8h h[2]; } u;
  u.h[0] = *(const v8h*)(p);
  u.h[1] = *(const v8h*)(p + 16);
  return u.v;
}
__device__ __forceinline__ v16h ld_bfrag(const _Float16* p) {
  // B 32x16 f16 fragment: halfs j=0..7 at k=8g.., j=8..15 at k=16+8g..
  union { v16h v; v8h h[2]; } u;
  u.h[0] = *(const v8h*)(p);
  u.h[1] = *(const v8h*)(p + 16);
  return u.v;
}

// One-block kernel: build padded, transposed, f16 [Wk|Wq|Wv]^T (112x160) in d_ws.
__global__ void __launch_bounds__(256) prep_w_kernel(const float* __restrict__ Wk,
                                                     const float* __restrict__ Wq,
                                                     const float* __restrict__ Wv,
                                                     _Float16* __restrict__ wout) {
  for (int i = threadIdx.x; i < WROWS * CP; i += 256) {
    int n = i / CP, k = i - (i / CP) * CP;
    float v = 0.0f;
    if (n < 3 * H_DIM && k < C_EMB) {
      if (n < H_DIM)            v = Wk[k * H_DIM + n];
      else if (n < 2 * H_DIM)   v = Wq[k * H_DIM + (n - H_DIM)];
      else                      v = Wv[k * H_DIM + (n - 2 * H_DIM)];
    }
    wout[i] = (_Float16)v;
  }
}

__global__ void __launch_bounds__(256) attn_head_kernel(
    const float* __restrict__ x,
    const float* __restrict__ Wk,
    const float* __restrict__ Wq,
    const float* __restrict__ Wv,
    const _Float16* __restrict__ wprep,
    int use_prep,
    float* __restrict__ out)
{
  extern __shared__ _Float16 sh[];
  const int b    = blockIdx.x;
  const int tid  = threadIdx.x;
  const int wv   = tid >> 5;    // wave id == M row-tile (0..7)
  const int lane = tid & 31;
  const int n16  = lane & 15;
  const int g    = lane >> 4;

  _Float16* xs  = sh + XS_OFF;
  _Float16* wt  = sh + WT_OFF;
  _Float16* kbf = sh + KB_OFF;
  _Float16* qbf = sh + QB_OFF;
  _Float16* vtb = sh + VT_OFF;
  _Float16* pb  = sh + PB_OFF;

  // ---- zero pad-sensitive LDS regions ----
  {
    uint32_t* z1 = (uint32_t*)xs;            // 20480 halfs
    for (int i = tid; i < 20480 / 2; i += 256) z1[i] = 0u;
    uint32_t* z2 = (uint32_t*)kbf;           // K+Q+V^T = 24576 halfs
    for (int i = tid; i < 24576 / 2; i += 256) z2[i] = 0u;
  }
  __syncthreads();

  // ---- stage x (f32 -> f16), fully coalesced float4 loads ----
  {
    const float4* xb = (const float4*)(x + (size_t)b * (T_SEQ * C_EMB));
    for (int i = tid; i < (T_SEQ * C_EMB) / 4; i += 256) {
      float4 q4 = xb[i];
      int f = i * 4;
      int t = f / C_EMB, c = f - t * C_EMB;
      v4h h4;
      h4[0] = (_Float16)q4.x; h4[1] = (_Float16)q4.y;
      h4[2] = (_Float16)q4.z; h4[3] = (_Float16)q4.w;
      *(v4h*)(xs + t * CP + c) = h4;
    }
  }
  // ---- stage weights ----
  if (use_prep) {
    const uint4* src = (const uint4*)wprep;
    uint4* dst = (uint4*)wt;
    for (int i = tid; i < (WROWS * CP) / 8; i += 256) dst[i] = src[i];
  } else {
    for (int i = tid; i < WROWS * CP; i += 256) {
      int n = i / CP, k = i - (i / CP) * CP;
      float v = 0.0f;
      if (n < 3 * H_DIM && k < C_EMB) {
        if (n < H_DIM)            v = Wk[k * H_DIM + n];
        else if (n < 2 * H_DIM)   v = Wq[k * H_DIM + (n - H_DIM)];
        else                      v = Wv[k * H_DIM + (n - 2 * H_DIM)];
      }
      wt[i] = (_Float16)v;
    }
  }
  __syncthreads();

  // ============ Phase 1: fused [K|Q|V] = x @ [Wk|Wq|Wv]  (128x112, K=160) ============
  {
    v16h afr[5];
#pragma unroll
    for (int kb = 0; kb < 5; ++kb)
      afr[kb] = ld_afrag(xs + (wv * 16 + n16) * CP + kb * 32 + 8 * g);

#pragma unroll
    for (int nt = 0; nt < 7; ++nt) {
      v8f acc = {0.f, 0.f, 0.f, 0.f, 0.f, 0.f, 0.f, 0.f};
#pragma unroll
      for (int kb = 0; kb < 5; ++kb) {
        v16h bfr = ld_bfrag(wt + (nt * 16 + n16) * CP + kb * 32 + 8 * g);
        acc = __builtin_amdgcn_wmma_f32_16x16x32_f16(false, afr[kb], false, bfr,
                                                     (short)0, acc, false, false);
      }
      // route D tile: lane column c, rows t0..t0+7
      int c  = nt * 16 + n16;
      int t0 = wv * 16 + 8 * g;
      if (c < H_DIM) {
#pragma unroll
        for (int i = 0; i < 8; ++i) kbf[(t0 + i) * 64 + c] = (_Float16)acc[i];
      } else if (c < 2 * H_DIM) {
#pragma unroll
        for (int i = 0; i < 8; ++i) qbf[(t0 + i) * 64 + (c - H_DIM)] = (_Float16)acc[i];
      } else if (c < 3 * H_DIM) {
        _Float16* dp = vtb + (c - 2 * H_DIM) * TP + t0;  // V^T: contiguous in t
#pragma unroll
        for (int i = 0; i < 8; ++i) dp[i] = (_Float16)acc[i];
      }
    }
  }
  __syncthreads();

  // ============ Phase 2: S = K @ Q^T (causal tiles only), softmax in registers ============
  v8f st[8];
  {
    v16h ka[2];
#pragma unroll
    for (int kb = 0; kb < 2; ++kb)
      ka[kb] = ld_afrag(kbf + (wv * 16 + n16) * 64 + kb * 32 + 8 * g);

#pragma unroll
    for (int nt = 0; nt < 8; ++nt) {
      if (nt <= wv) {   // wave-uniform branch: EXEC stays all-1s for WMMA
        v8f acc = {0.f, 0.f, 0.f, 0.f, 0.f, 0.f, 0.f, 0.f};
#pragma unroll
        for (int kb = 0; kb < 2; ++kb) {
          v16h bfr = ld_bfrag(qbf + (nt * 16 + n16) * 64 + kb * 32 + 8 * g);
          acc = __builtin_amdgcn_wmma_f32_16x16x32_f16(false, ka[kb], false, bfr,
                                                       (short)0, acc, false, false);
        }
        st[nt] = acc;
      }
    }

    const float scale   = 1.0f / 12.0f;   // C**-0.5, C = n_embd = 144
    const float NEG_BIG = -3.0e38f;
    float rmax[8];
#pragma unroll
    for (int i = 0; i < 8; ++i) rmax[i] = NEG_BIG;
#pragma unroll
    for (int nt = 0; nt < 8; ++nt) {
      if (nt <= wv) {
        int cc = nt * 16 + n16;
#pragma unroll
        for (int i = 0; i < 8; ++i) {
          int r = wv * 16 + 8 * g + i;
          float v = st[nt][i] * scale;
          if (cc > r || cc >= T_SEQ) v = NEG_BIG;   // causal + column padding mask
          st[nt][i] = v;
          rmax[i] = fmaxf(rmax[i], v);
        }
      }
    }
    // row lives in a 16-lane group in the D layout: reduce with xor shuffles 1..8
#pragma unroll
    for (int off = 1; off < 16; off <<= 1)
#pragma unroll
      for (int i = 0; i < 8; ++i)
        rmax[i] = fmaxf(rmax[i], __shfl_xor(rmax[i], off, 32));

    float rsum[8];
#pragma unroll
    for (int i = 0; i < 8; ++i) rsum[i] = 0.f;
#pragma unroll
    for (int nt = 0; nt < 8; ++nt) {
      if (nt <= wv) {
#pragma unroll
        for (int i = 0; i < 8; ++i) {
          float e = __expf(st[nt][i] - rmax[i]);
          st[nt][i] = e;
          rsum[i] += e;
        }
      }
    }
#pragma unroll
    for (int off = 1; off < 16; off <<= 1)
#pragma unroll
      for (int i = 0; i < 8; ++i)
        rsum[i] += __shfl_xor(rsum[i], off, 32);
#pragma unroll
    for (int i = 0; i < 8; ++i) rsum[i] = 1.0f / rsum[i];

    // write P (f16) — each wave owns its 16 rows entirely
#pragma unroll
    for (int nt = 0; nt < 8; ++nt) {
#pragma unroll
      for (int i = 0; i < 8; ++i) {
        int r = wv * 16 + 8 * g + i;
        _Float16 pv = (nt <= wv) ? (_Float16)(st[nt][i] * rsum[i]) : (_Float16)0.0f;
        pb[r * TP + nt * 16 + n16] = pv;
      }
    }
  }
  __syncthreads();

  // ============ Phase 3: O = P @ V (128x48 used cols, K=128) ============
  {
    v16h pa[4];
#pragma unroll
    for (int kb = 0; kb < 4; ++kb)
      pa[kb] = ld_afrag(pb + (wv * 16 + n16) * TP + kb * 32 + 8 * g);

    float* ob = out + (size_t)b * (T_SEQ * H_DIM);
#pragma unroll
    for (int ht = 0; ht < 3; ++ht) {
      v8f acc = {0.f, 0.f, 0.f, 0.f, 0.f, 0.f, 0.f, 0.f};
#pragma unroll
      for (int kb = 0; kb < 4; ++kb) {
        v16h bfr = ld_bfrag(vtb + (ht * 16 + n16) * TP + kb * 32 + 8 * g);
        acc = __builtin_amdgcn_wmma_f32_16x16x32_f16(false, pa[kb], false, bfr,
                                                     (short)0, acc, false, false);
      }
      int h = ht * 16 + n16;
      if (h < H_DIM) {
#pragma unroll
        for (int i = 0; i < 8; ++i) {
          int t = wv * 16 + 8 * g + i;
          if (t < T_SEQ) ob[t * H_DIM + h] = acc[i];   // 16 lanes = 16 consecutive floats
        }
      }
    }
  }
}

extern "C" void kernel_launch(void* const* d_in, const int* in_sizes, int n_in,
                              void* d_out, int out_size, void* d_ws, size_t ws_size,
                              hipStream_t stream) {
  (void)in_sizes; (void)n_in; (void)out_size;
  const float* x  = (const float*)d_in[0];
  const float* Wk = (const float*)d_in[1];
  const float* Wq = (const float*)d_in[2];
  const float* Wv = (const float*)d_in[3];
  float* out = (float*)d_out;

  const size_t smem = (size_t)SH_HALFS * sizeof(_Float16);   // 125952 B < 320 KB/WGP
  // opt in to >64KB dynamic LDS (no-op / harmless on AMD if unsupported)
  (void)hipFuncSetAttribute((const void*)attn_head_kernel,
                            hipFuncAttributeMaxDynamicSharedMemorySize, (int)smem);

  const size_t wbytes = (size_t)WROWS * CP * sizeof(_Float16);  // 35840
  int use_prep = (d_ws != nullptr && ws_size >= wbytes) ? 1 : 0;
  if (use_prep)
    prep_w_kernel<<<1, 256, 0, stream>>>(Wk, Wq, Wv, (_Float16*)d_ws);

  attn_head_kernel<<<B_BATCH, 256, smem, stream>>>(x, Wk, Wq, Wv,
                                                   (const _Float16*)d_ws, use_prep, out);
}
